// RAN_40037685134208
// MI455X (gfx1250) — matrix operand order repaired and checked by reading, not compiled, not run on hardware
//
#include <hip/hip_runtime.h>
#include <math.h>

// Problem constants (match reference)
#define T_SEQ 512
#define BATCH 64
#define DIN   1024
#define DH    1024
#define MTOT  (T_SEQ * BATCH)   // 32768 rows for the batched projections
#define NWG_REC 64              // workgroups cooperating on the recurrence

typedef __attribute__((ext_vector_type(16))) __bf16       v16bf;
typedef __attribute__((ext_vector_type(4)))  __bf16       v4bf;
typedef __attribute__((ext_vector_type(8)))  float        v8f;
typedef __attribute__((ext_vector_type(8)))  unsigned int v8u;
typedef __attribute__((ext_vector_type(4)))  int          v4i;
typedef __attribute__((ext_vector_type(8)))  int          v8i;

// LDS aperture: low 32 bits of a generic shared pointer are the LDS offset.
static __device__ __forceinline__ unsigned lds_off32(const void* p) {
  return (unsigned)(unsigned long long)p;
}

// Async DMA: global (16B per lane) -> LDS, tracked by ASYNCcnt.
static __device__ __forceinline__ void async_g2l_b128(unsigned lds_dst,
                                                      const void* gsrc) {
  asm volatile("global_load_async_to_lds_b128 %0, %1, off"
               :
               : "v"(lds_dst), "v"(gsrc)
               : "memory");
}

// TDM: DMA a 2-D tile (16 rows x 1024 bf16, row stride 1024) global -> LDS,
// with LDS padding of 2 DWORDs every 256 DWORDs (row stride 516 DWORDs).
// D# group0 (4 SGPRs) + group1 (8 SGPRs); groups 2/3 unused (2-D tensor).
static __device__ __forceinline__ void tdm_load_w16(unsigned lds_byte_off,
                                                    const void* gsrc) {
  unsigned long long ga = (unsigned long long)gsrc;
  v4i g0;
  g0[0] = __builtin_amdgcn_readfirstlane(1);                 // count=1, user D#
  g0[1] = __builtin_amdgcn_readfirstlane((int)lds_byte_off); // lds_addr
  g0[2] = __builtin_amdgcn_readfirstlane((int)(ga & 0xFFFFFFFFu));
  g0[3] = __builtin_amdgcn_readfirstlane(
      (int)(((ga >> 32) & 0x01FFFFFFu) | 0x80000000u));      // addr hi | type=2
  v8i g1;
  // data_size=2B (1<<16), pad_enable (1<<20), pad_interval=7 (256 DW, 7<<22),
  // pad_amount=2 DW (1<<25); workgroup_mask=0 (not in cluster)
  g1[0] = __builtin_amdgcn_readfirstlane((int)((1u << 16) | (1u << 20) |
                                               (7u << 22) | (1u << 25)));
  g1[1] = __builtin_amdgcn_readfirstlane((int)(1024u << 16)); // tensor_dim0=1024
  g1[2] = __builtin_amdgcn_readfirstlane((int)(16u << 16));   // tensor_dim1=16
  g1[3] = __builtin_amdgcn_readfirstlane((int)(1024u << 16)); // tile_dim0=1024
  g1[4] = __builtin_amdgcn_readfirstlane(16);                 // tile_dim1=16
  g1[5] = __builtin_amdgcn_readfirstlane(1024);               // dim0_stride=1024
  g1[6] = __builtin_amdgcn_readfirstlane(0);
  g1[7] = __builtin_amdgcn_readfirstlane(0);
  asm volatile("tensor_load_to_lds %0, %1" : : "s"(g0), "s"(g1) : "memory");
}

static __device__ __forceinline__ void wait_tensorcnt0() {
#if defined(__has_builtin) && __has_builtin(__builtin_amdgcn_s_wait_tensorcnt)
  __builtin_amdgcn_s_wait_tensorcnt(0);
#else
  asm volatile("s_wait_tensorcnt 0x0" ::: "memory");
#endif
}

// ---------------------------------------------------------------------------
// fp32 -> bf16 conversion (vectorized: float4 in, 4x bf16 out)
// ---------------------------------------------------------------------------
__global__ __launch_bounds__(256) void k_cvt4(const float4* __restrict__ in,
                                              __bf16* __restrict__ out, int n4) {
  int i = blockIdx.x * blockDim.x + threadIdx.x;
  if (i < n4) {
    float4 f = in[i];
    v4bf o;
    o[0] = (__bf16)f.x; o[1] = (__bf16)f.y;
    o[2] = (__bf16)f.z; o[3] = (__bf16)f.w;
    *(v4bf*)(out + 4 * (size_t)i) = o;
  }
}

__global__ __launch_bounds__(256) void k_zero(unsigned int* __restrict__ p, int n) {
  int i = blockIdx.x * blockDim.x + threadIdx.x;
  if (i < n) p[i] = 0u;
}

// ---------------------------------------------------------------------------
// Fused batched projections (ctilde -> d_out, ix/fx -> ws).
// WG tile 64(M) x 64(N); 8 waves = (nhalf, mtile); A tiles double-buffered in
// LDS via global_load_async_to_lds_b128, one barrier per k-step.
// ---------------------------------------------------------------------------
#define A_STRIDE 20  // uints per 32-bf16 row: 80B (16B aligned, conflict-free)

__global__ __launch_bounds__(256) void k_prep(const __bf16* __restrict__ xb,
                                              const __bf16* __restrict__ wcx,
                                              const __bf16* __restrict__ wix,
                                              const __bf16* __restrict__ wfx,
                                              const float* __restrict__ bcx,
                                              const float* __restrict__ bix,
                                              const float* __restrict__ bfx,
                                              float* __restrict__ out_c,
                                              float* __restrict__ out_i,
                                              float* __restrict__ out_f) {
  __shared__ unsigned int ldsA[2][64 * A_STRIDE];

  const int tid  = threadIdx.x;
  const int lane = tid & 31;
  const int wv   = tid >> 5;
  const int mt   = wv & 3;
  const int nh   = wv >> 2;
  const int lrow = lane & 15;
  const int lhi  = lane >> 4;

  const int mbase = blockIdx.x * 64;
  const int nbase = blockIdx.y * 64 + nh * 32;

  const int srow  = tid >> 2;
  const int skseg = (tid & 3) * 8;
  const __bf16* sbase = xb + (size_t)(mbase + srow) * DIN + skseg;
  const unsigned sdst0 = lds_off32(&ldsA[0][srow * A_STRIDE + (skseg >> 1)]);
  const unsigned sdst1 = lds_off32(&ldsA[1][srow * A_STRIDE + (skseg >> 1)]);

  // named accumulators: keep them pinned (no phi copies)
  v8f aC0 = (v8f){}, aC1 = (v8f){};
  v8f aI0 = (v8f){}, aI1 = (v8f){};
  v8f aF0 = (v8f){}, aF1 = (v8f){};

  auto stepf = [&](const unsigned int* __restrict__ buf, int kbase) {
    v8u au;
    const unsigned int* base = buf + (mt * 16 + lrow) * A_STRIDE;
    #pragma unroll
    for (int v = 0; v < 8; ++v) {
      int idx = lhi * 8 + v;
      au[v] = base[idx];
    }
    v16bf a = __builtin_bit_cast(v16bf, au);
    const size_t o0 = (size_t)(nbase + lrow) * DIN + kbase + lhi * 16;
    const size_t o1 = o0 + (size_t)16 * DIN;
    aC0 = __builtin_amdgcn_wmma_f32_16x16x32_bf16(
        false, a, false, *(const v16bf*)(wcx + o0), (short)0, aC0, false, false);
    aI0 = __builtin_amdgcn_wmma_f32_16x16x32_bf16(
        false, a, false, *(const v16bf*)(wix + o0), (short)0, aI0, false, false);
    aF0 = __builtin_amdgcn_wmma_f32_16x16x32_bf16(
        false, a, false, *(const v16bf*)(wfx + o0), (short)0, aF0, false, false);
    aC1 = __builtin_amdgcn_wmma_f32_16x16x32_bf16(
        false, a, false, *(const v16bf*)(wcx + o1), (short)0, aC1, false, false);
    aI1 = __builtin_amdgcn_wmma_f32_16x16x32_bf16(
        false, a, false, *(const v16bf*)(wix + o1), (short)0, aI1, false, false);
    aF1 = __builtin_amdgcn_wmma_f32_16x16x32_bf16(
        false, a, false, *(const v16bf*)(wfx + o1), (short)0, aF1, false, false);
  };

  // prologue: stage tile 0 into buffer 0
  async_g2l_b128(sdst0, sbase);

  #pragma unroll 1
  for (int kt = 0; kt < DIN / 32; kt += 2) {
    // even step: compute buf0, prefetch tile kt+1 into buf1
    asm volatile("s_wait_asynccnt 0x0" ::: "memory");
    __syncthreads();                    // publish buf0; buf1 reads all done
    async_g2l_b128(sdst1, sbase + (kt + 1) * 32);
    stepf(&ldsA[0][0], kt * 32);
    // odd step: compute buf1, prefetch tile kt+2 into buf0
    asm volatile("s_wait_asynccnt 0x0" ::: "memory");
    __syncthreads();
    if (kt + 2 < DIN / 32)
      async_g2l_b128(sdst0, sbase + (kt + 2) * 32);
    stepf(&ldsA[1][0], (kt + 1) * 32);
  }

  // Store with bias. C layout: VGPR r -> M = r + 8*lhi, lane%16 -> N.
  {
    int n0 = nbase + lrow;
    int n1 = n0 + 16;
    float bc0 = bcx[n0], bi0 = bix[n0], bf0 = bfx[n0];
    float bc1 = bcx[n1], bi1 = bix[n1], bf1 = bfx[n1];
    #pragma unroll
    for (int r = 0; r < 8; ++r) {
      size_t row = (size_t)(mbase + mt * 16 + lhi * 8 + r) * DH;
      out_c[row + n0] = aC0[r] + bc0;
      out_i[row + n0] = aI0[r] + bi0;
      out_f[row + n0] = aF0[r] + bf0;
      out_c[row + n1] = aC1[r] + bc1;
      out_i[row + n1] = aI1[r] + bi1;
      out_f[row + n1] = aF1[r] + bf1;
    }
  }
}

// ---------------------------------------------------------------------------
// Recurrence: 64 WGs, each owns 16 hidden columns. Weight slices (16x1024 bf16
// per gate) are DMA'd once into LDS by the TDM (tensor_load_to_lds) and stay
// resident for all 512 steps. Split-K dual accumulators for matrix-unit ILP.
// Grid-wide sync per step via per-t atomic counters.
// ---------------------------------------------------------------------------
#define W_STRIDE 516  // DWORDs per 1024-bf16 row after TDM padding (512 + 2*2)

__global__ __launch_bounds__(256) void k_recur(const float* __restrict__ h0,
                                               const __bf16* __restrict__ wic,
                                               const __bf16* __restrict__ wfc,
                                               const float* __restrict__ bic,
                                               const float* __restrict__ bfc,
                                               const float* __restrict__ ixp,
                                               const float* __restrict__ fxp,
                                               float* __restrict__ out,
                                               float* __restrict__ hlast,
                                               unsigned int* __restrict__ cnt) {
  __shared__ unsigned int ldsW[2][16 * W_STRIDE];  // padded weight tiles (i,f)
  __shared__ float ldsP[2][BATCH][16];             // pre-activations

  const int tid   = threadIdx.x;
  const int lane  = tid & 31;
  const int wv    = tid >> 5;
  const int g     = wv >> 2;      // 0 = i gate, 1 = f gate
  const int mt    = wv & 3;       // 16-row batch tile
  const int lrow  = lane & 15;
  const int lhi   = lane >> 4;
  const int nbase = blockIdx.x * 16;

  // Stage both weight slices into LDS via TDM (wave 0 issues; done-signal is
  // per instruction, EXEC ignored).
  if (wv == 0) {
    tdm_load_w16(lds_off32(&ldsW[0][0]), wic + (size_t)nbase * DH);
    tdm_load_w16(lds_off32(&ldsW[1][0]), wfc + (size_t)nbase * DH);
    wait_tensorcnt0();
  }
  __syncthreads();

  const unsigned int* wls = &ldsW[g][0];

  // per-thread elementwise coordinates + preloaded biases
  const int br = tid >> 2;
  const int j0 = (tid & 3) * 4;
  float bicv[4], bfcv[4];
  #pragma unroll
  for (int jj = 0; jj < 4; ++jj) {
    bicv[jj] = bic[nbase + j0 + jj];
    bfcv[jj] = bfc[nbase + j0 + jj];
  }

  for (int t = 0; t < T_SEQ; ++t) {
    const float* hprev =
        (t == 0) ? h0 : (out + (size_t)(t - 1) * BATCH * DH);

    // --- GEMM: 16x16 tile of pre-activation, K = 1024, two indep. chains ---
    v8f acc0 = (v8f){};
    v8f acc1 = (v8f){};
    const float* hrow = hprev + (size_t)(mt * 16 + lrow) * DH;
    const unsigned int* wrow = wls + lrow * W_STRIDE;
    #pragma unroll 1
    for (int kt = 0; kt < DH / 64; ++kt) {
      #pragma unroll
      for (int half = 0; half < 2; ++half) {
        const int kbase = kt * 64 + half * 32;
        v16bf a;
        #pragma unroll
        for (int v = 0; v < 8; ++v) {
          int k0 = kbase + lhi * 16 + 2 * v;
          float2 f2 = *(const float2*)(hrow + k0);
          a[2 * v]     = (__bf16)f2.x;
          a[2 * v + 1] = (__bf16)f2.y;
        }
        // B fragment from padded LDS tile (8B-aligned pieces)
        const int kk = kbase + lhi * 16;
        const int d  = (kk >> 1) + ((kk >> 8) & 2);  // +2 DW pad past k=512
        const unsigned int* p = wrow + d;
        uint2 q0 = *(const uint2*)(p);
        uint2 q1 = *(const uint2*)(p + 2);
        uint2 q2 = *(const uint2*)(p + 4);
        uint2 q3 = *(const uint2*)(p + 6);
        v8u bu = {q0.x, q0.y, q1.x, q1.y, q2.x, q2.y, q3.x, q3.y};
        v16bf b = __builtin_bit_cast(v16bf, bu);
        if (half == 0)
          acc0 = __builtin_amdgcn_wmma_f32_16x16x32_bf16(
              false, a, false, b, (short)0, acc0, false, false);
        else
          acc1 = __builtin_amdgcn_wmma_f32_16x16x32_bf16(
              false, a, false, b, (short)0, acc1, false, false);
      }
    }
    v8f acc = acc0 + acc1;
    #pragma unroll
    for (int r = 0; r < 8; ++r)
      ldsP[g][mt * 16 + lhi * 8 + r][lrow] = acc[r];
    __syncthreads();

    // --- Gate math + in-place h write (ctilde_t lives in out[t]) ---
    {
      const size_t base = (size_t)t * BATCH * DH + (size_t)br * DH + nbase;
      #pragma unroll
      for (int jj = 0; jj < 4; ++jj) {
        int j = j0 + jj;
        float ipre = ldsP[0][br][j] + bicv[jj] + ixp[base + j];
        float fpre = ldsP[1][br][j] + bfcv[jj] + fxp[base + j];
        float ct = out[base + j];
        float hp = hprev[(size_t)br * DH + nbase + j];
        float iv = 1.0f / (1.0f + __expf(-ipre));
        float fv = 1.0f / (1.0f + __expf(-fpre));
        out[base + j] = tanhf(iv * ct + fv * hp);
      }
    }
    __threadfence();
    __syncthreads();

    // --- grid-wide step barrier (per-t counter, release/acquire) ---
    if (tid == 0) {
      __hip_atomic_fetch_add(&cnt[t], 1u, __ATOMIC_RELEASE,
                             __HIP_MEMORY_SCOPE_AGENT);
      while (__hip_atomic_load(&cnt[t], __ATOMIC_ACQUIRE,
                               __HIP_MEMORY_SCOPE_AGENT) < (unsigned)NWG_REC)
        __builtin_amdgcn_s_sleep(1);
    }
    __syncthreads();
  }

  // h_last = out[T-1]
  {
    const float* hfin = out + (size_t)(T_SEQ - 1) * BATCH * DH;
    int idx = blockIdx.x * 1024 + tid * 4;
    float4 v = *(const float4*)(hfin + idx);
    *(float4*)(hlast + idx) = v;
  }
}

// ---------------------------------------------------------------------------
// Launch
// ---------------------------------------------------------------------------
extern "C" void kernel_launch(void* const* d_in, const int* in_sizes, int n_in,
                              void* d_out, int out_size, void* d_ws,
                              size_t ws_size, hipStream_t stream) {
  const float* x    = (const float*)d_in[0];
  const float* h0   = (const float*)d_in[1];
  const float* wcx  = (const float*)d_in[2];
  const float* wic  = (const float*)d_in[3];
  const float* wix  = (const float*)d_in[4];
  const float* wfc  = (const float*)d_in[5];
  const float* wfx  = (const float*)d_in[6];
  const float* bcx  = (const float*)d_in[7];
  const float* bic  = (const float*)d_in[8];
  const float* bix  = (const float*)d_in[9];
  const float* bfc  = (const float*)d_in[10];
  const float* bfx  = (const float*)d_in[11];

  float* out   = (float*)d_out;                       // [T,B,H] (ctilde first)
  float* hlast = out + (size_t)MTOT * DH;             // [B,H]

  // Workspace layout
  const size_t XB_BYTES = (size_t)MTOT * DIN * 2;     // bf16 x     (64 MB)
  const size_t W_BYTES  = (size_t)DH * DIN * 2;       // bf16 w     (2 MB each)
  const size_t GX_BYTES = (size_t)MTOT * DH * 4;      // f32 ix/fx  (128 MB each)
  char* ws = (char*)d_ws;
  __bf16* xb  = (__bf16*)(ws);
  __bf16* wcb = (__bf16*)(ws + XB_BYTES + 0 * W_BYTES);
  __bf16* wib = (__bf16*)(ws + XB_BYTES + 1 * W_BYTES);  // w_ic
  __bf16* wxb = (__bf16*)(ws + XB_BYTES + 2 * W_BYTES);  // w_ix
  __bf16* wfb = (__bf16*)(ws + XB_BYTES + 3 * W_BYTES);  // w_fc
  __bf16* wyb = (__bf16*)(ws + XB_BYTES + 4 * W_BYTES);  // w_fx
  float* ixp  = (float*)(ws + XB_BYTES + 5 * W_BYTES);
  float* fxp  = (float*)(ws + XB_BYTES + 5 * W_BYTES + GX_BYTES);
  unsigned int* cnt =
      (unsigned int*)(ws + XB_BYTES + 5 * W_BYTES + 2 * GX_BYTES);

  // bf16 conversions
  {
    int n4 = (MTOT * DIN) / 4;
    k_cvt4<<<(n4 + 255) / 256, 256, 0, stream>>>((const float4*)x, xb, n4);
    int w4 = (DH * DIN) / 4;
    int blk = (w4 + 255) / 256;
    k_cvt4<<<blk, 256, 0, stream>>>((const float4*)wcx, wcb, w4);
    k_cvt4<<<blk, 256, 0, stream>>>((const float4*)wic, wib, w4);
    k_cvt4<<<blk, 256, 0, stream>>>((const float4*)wix, wxb, w4);
    k_cvt4<<<blk, 256, 0, stream>>>((const float4*)wfc, wfb, w4);
    k_cvt4<<<blk, 256, 0, stream>>>((const float4*)wfx, wyb, w4);
  }

  // zero the per-step barrier counters (must happen every call)
  k_zero<<<2, 256, 0, stream>>>(cnt, T_SEQ);

  // fused batched projections (ctilde -> d_out, ix/fx -> ws)
  k_prep<<<dim3(MTOT / 64, DH / 64), 256, 0, stream>>>(
      xb, wcb, wxb, wyb, bcx, bix, bfx, out, ixp, fxp);

  // sequential recurrence
  k_recur<<<NWG_REC, 256, 0, stream>>>(h0, wib, wfb, bic, bfc, ixp, fxp, out,
                                       hlast, cnt);
}
